// SimpleBICEPLayer_22445499089475
// MI455X (gfx1250) — matrix operand order repaired and checked by reading, not compiled, not run on hardware
//
#include <hip/hip_runtime.h>
#include <hip/hip_bf16.h>

typedef __attribute__((ext_vector_type(16))) _Float16 v16h;
typedef __attribute__((ext_vector_type(8)))  _Float16 v8h;
typedef __attribute__((ext_vector_type(8)))  float    v8f;

#define BB     8192
#define DIN    512
#define DOUT   1024
#define NP     256
#define NS     50

// ---------------------------------------------------------------------------
// Kernel 1: convert W_agg [DOUT, NP] fp32 -> fp16 (workspace)
// ---------------------------------------------------------------------------
__global__ void cvt_wagg_f16(const float* __restrict__ w, _Float16* __restrict__ w16) {
    int i = blockIdx.x * blockDim.x + threadIdx.x;
    w16[i] = (_Float16)w[i];
}

// ---------------------------------------------------------------------------
// Kernel 2: per-sample feedback + endpoint reduction (the 419 MB stream).
// One block (256 threads) per batch sample b:
//   - block-reduce x[b]·W_fb (512 elems, 2 per thread)
//   - sigmoid -> fb ; scale[s] = 0.5 + fb*exp(-0.1*s*dt) into LDS
//   - thread p streams noise[b,p,0..49] (float2 loads, rows are 8B aligned)
//     endpoint = sqrt(dt) * sum_s scale[s]*noise[b,p,s], stored as f16
// ---------------------------------------------------------------------------
__global__ void __launch_bounds__(256)
fb_endpoints(const float* __restrict__ x, const float* __restrict__ wfb,
             const float* __restrict__ bfb, const float* __restrict__ noise,
             _Float16* __restrict__ e16) {
    __shared__ float red[256];
    __shared__ float sscale[NS];

    const int b = blockIdx.x;
    const int t = threadIdx.x;

    const float* xr = x + (size_t)b * DIN;
    red[t] = xr[t] * wfb[t] + xr[t + 256] * wfb[t + 256];
    __syncthreads();
    #pragma unroll
    for (int s = 128; s > 0; s >>= 1) {
        if (t < s) red[t] += red[t + s];
        __syncthreads();
    }
    if (t < NS) {
        float z  = red[0] + bfb[0];
        float fb = 1.0f / (1.0f + __expf(-z));                 // sigmoid
        sscale[t] = 0.5f + fb * __expf(-0.1f * (0.02f * (float)t)); // dt = 1/50
    }
    __syncthreads();

    // stream this thread's path: 50 contiguous floats (25 float2, 8B aligned)
    const float2* nrow = (const float2*)(noise + ((size_t)b * NP + t) * NS);
    float acc = 0.0f;
    #pragma unroll
    for (int j = 0; j < NS / 2; ++j) {
        float2 v = nrow[j];
        acc += v.x * sscale[2 * j] + v.y * sscale[2 * j + 1];
    }
    e16[(size_t)b * NP + t] = (_Float16)(acc * 0.14142135623730951f); // sqrt(dt)
}

// ---------------------------------------------------------------------------
// Kernel 3: out[8192,1024] = endpoints[8192,256] @ W_agg^T + b_agg
// via v_wmma_f32_16x16x32_f16. One wave per 16x16 tile, K=256 -> 8 WMMAs.
//
// A fragment (16x32 f16, ISA 7.12.2): lane(M=l15, hi):
//   elems 0..7  = A[M][kb + 8*hi      .. +7]   (one b128 load)
//   elems 8..15 = A[M][kb + 16 + 8*hi .. +7]   (one b128 load)
// B fragment (32x16 f16, per B-matrix tables): lane(N=l15, hi):
//   elems 0..7  = W_agg[N][kb + 8*hi      .. +7]   (one b128 load)
//   elems 8..15 = W_agg[N][kb + 16 + 8*hi .. +7]   (one b128 load)
// C/D: VGPR r -> row M = m0 + 8*hi + r, col N = n0 + l15.
// ---------------------------------------------------------------------------
__global__ void __launch_bounds__(256)
agg_gemm_wmma(const _Float16* __restrict__ E16, const _Float16* __restrict__ W16,
              const float* __restrict__ bagg, float* __restrict__ out) {
    const int wave = threadIdx.x >> 5;
    const int lane = threadIdx.x & 31;
    const int tile = blockIdx.x * 8 + wave;      // (8192/16)*(1024/16) = 32768 tiles
    const int mt   = tile >> 6;                  // 64 n-tiles per m row
    const int nt   = tile & 63;
    const int m0   = mt * 16;
    const int n0   = nt * 16;
    const int l15  = lane & 15;
    const int hi   = lane >> 4;

    const _Float16* Arow = E16 + (size_t)(m0 + l15) * NP;  // A row M
    const _Float16* Brow = W16 + (size_t)(n0 + l15) * NP;  // W_agg row = B column N

    v8f c = {};
    #pragma unroll
    for (int kb = 0; kb < NP; kb += 32) {
        v16h a, bm;
        ((v8h*)&a)[0]  = *(const v8h*)(Arow + kb + hi * 8);        // 16B aligned
        ((v8h*)&a)[1]  = *(const v8h*)(Arow + kb + 16 + hi * 8);
        ((v8h*)&bm)[0] = *(const v8h*)(Brow + kb + hi * 8);
        ((v8h*)&bm)[1] = *(const v8h*)(Brow + kb + 16 + hi * 8);
        c = __builtin_amdgcn_wmma_f32_16x16x32_f16(
                /*neg_a=*/false, a, /*neg_b=*/false, bm,
                /*c_mod=*/(short)0, c, /*reuse_a=*/false, /*reuse_b=*/false);
    }

    const float bias = bagg[n0 + l15];
    float* orow = out + (size_t)(m0 + hi * 8) * DOUT + (n0 + l15);
    #pragma unroll
    for (int r = 0; r < 8; ++r)
        orow[(size_t)r * DOUT] = c[r] + bias;
}

// ---------------------------------------------------------------------------
extern "C" void kernel_launch(void* const* d_in, const int* in_sizes, int n_in,
                              void* d_out, int out_size, void* d_ws, size_t ws_size,
                              hipStream_t stream) {
    const float* x     = (const float*)d_in[0];  // [8192, 512]
    const float* wfb   = (const float*)d_in[1];  // [1, 512]
    const float* bfb   = (const float*)d_in[2];  // [1]
    const float* wagg  = (const float*)d_in[3];  // [1024, 256]
    const float* bagg  = (const float*)d_in[4];  // [1024]
    const float* noise = (const float*)d_in[5];  // [8192, 256, 50]
    float* out = (float*)d_out;                  // [8192, 1024] f32

    _Float16* e16 = (_Float16*)d_ws;                                   // 4 MB
    _Float16* w16 = (_Float16*)((char*)d_ws + (size_t)BB * NP * 2);    // +512 KB

    cvt_wagg_f16<<<(DOUT * NP) / 256, 256, 0, stream>>>(wagg, w16);
    fb_endpoints<<<BB, 256, 0, stream>>>(x, wfb, bfb, noise, e16);
    agg_gemm_wmma<<<(BB / 16) * (DOUT / 16) / 8, 256, 0, stream>>>(e16, w16, bagg, out);
}
